// GCN_89988154785840
// MI455X (gfx1250) — matrix operand order repaired and checked by reading, not compiled, not run on hardware
//
#include <hip/hip_runtime.h>

typedef _Float16 half16 __attribute__((ext_vector_type(16)));
typedef float    float8 __attribute__((ext_vector_type(8)));

#define N_IN  128
#define HIDN  256
#define AD    512   // A*D
#define ASP_A 8
#define ASP_D 64

// ---------------- column mean of feature: mean[c] = (1/n) sum_v feat[v][c] -----------
__global__ void col_mean_kernel(const float* __restrict__ feat, float* __restrict__ mean, int n) {
  __shared__ float red[256];
  int c = blockIdx.x;  // 0..127
  float s = 0.f;
  for (int v = threadIdx.x; v < n; v += 256) s += feat[(size_t)v * N_IN + c];
  red[threadIdx.x] = s;
  __syncthreads();
  for (int off = 128; off > 0; off >>= 1) {
    if ((int)threadIdx.x < off) red[threadIdx.x] += red[threadIdx.x + off];
    __syncthreads();
  }
  if (threadIdx.x == 0) mean[c] = red[0] / (float)n;
}

// ---------------- zero S / deg / mask (contiguous region of `count` dwords) ----------
__global__ void zero_kernel(float* __restrict__ p, int count) {
  int i = blockIdx.x * 256 + threadIdx.x;
  if (i < count) p[i] = 0.f;
}

// ---------------- mark nodes that appear in pairs ------------------------------------
__global__ void mask_kernel(const int* __restrict__ pairsFlat, int* __restrict__ mask, int m) {
  int i = blockIdx.x * 256 + threadIdx.x;
  if (i < m) mask[pairsFlat[i]] = 1;
}

// ---------------- fp32 -> fp16 weight copies -----------------------------------------
__global__ void cvt_kernel(const float* __restrict__ W1, const float* __restrict__ aspW,
                           _Float16* __restrict__ W1h, _Float16* __restrict__ aspWh) {
  int i = blockIdx.x * 256 + threadIdx.x;
  if (i < HIDN * N_IN) W1h[i] = (_Float16)W1[i];
  if (i < AD * HIDN)  aspWh[i] = (_Float16)aspW[i];
}

// ---------------- masked edge aggregation: S[dst] += feat[src]+mean; deg[dst] += 1 ---
__global__ void edge_kernel(const int* __restrict__ src, const int* __restrict__ dst,
                            const float* __restrict__ feat, const float* __restrict__ mean,
                            const int* __restrict__ mask, float* __restrict__ S,
                            float* __restrict__ deg, int E) {
  int e = blockIdx.x * 2 + (threadIdx.x >> 7);
  if (e >= E) return;
  int c = threadIdx.x & 127;
  int d = dst[e];
  if (!mask[d]) return;
  if (c == 0) unsafeAtomicAdd(&deg[d], 1.0f);
  int s = src[e];
  float val = feat[(size_t)s * N_IN + c] + mean[c];
  unsafeAtomicAdd(&S[(size_t)d * N_IN + c], val);
}

// ---------------- GEMM1: xh[i] = ((nf ⊙ S / deg)[pairs[i]]) @ W1^T + b1  (f16 out) ---
// 16 rows per block, 256 threads = 8 waves, each wave does two 16x16 N-tiles, K=128.
__global__ void gemm1_kernel(const int* __restrict__ pairsFlat,
                             const float* __restrict__ feat, const float* __restrict__ mean,
                             const float* __restrict__ S, const float* __restrict__ deg,
                             const _Float16* __restrict__ W1h, const float* __restrict__ b1,
                             _Float16* __restrict__ xh) {
  __shared__ _Float16 At[16][136];  // padded row stride
  int i0  = blockIdx.x * 16;
  int tid = threadIdx.x;
  for (int idx = tid; idx < 16 * N_IN; idx += 256) {
    int r = idx >> 7, c = idx & 127;
    int v = pairsFlat[i0 + r];
    float dg = deg[v];
    dg = dg > 0.f ? dg : 1.f;
    float xp = (feat[(size_t)v * N_IN + c] + mean[c]) * S[(size_t)v * N_IN + c] / dg;
    At[r][c] = (_Float16)xp;
  }
  __syncthreads();
  int lane = tid & 31, wave = tid >> 5;
  int row  = lane & 15, h = lane >> 4;
  for (int t = 0; t < 2; ++t) {
    int n0  = (wave * 2 + t) * 16;
    int col = n0 + row;  // N index owned by this lane for B/C/D
    float8 acc = {0.f, 0.f, 0.f, 0.f, 0.f, 0.f, 0.f, 0.f};
    for (int k0 = 0; k0 < N_IN; k0 += 32) {
      half16 a;
#pragma unroll
      for (int vv = 0; vv < 8; ++vv) {  // A-layout (matches B): K = 16*h + 2*v
        int k = k0 + (h << 4) + (vv << 1);
        a[2 * vv]     = At[row][k];
        a[2 * vv + 1] = At[row][k + 1];
      }
      half16 b = *(const half16*)(W1h + (size_t)col * N_IN + k0 + h * 16);
      acc = __builtin_amdgcn_wmma_f32_16x16x32_f16(false, a, false, b, (short)0, acc, false, false);
    }
    float bias = b1[col];
#pragma unroll
    for (int r = 0; r < 8; ++r) {
      int m = r + 8 * h;  // D layout: M = r + 8*(lane>=16)
      xh[(size_t)(i0 + m) * HIDN + col] = (_Float16)(acc[r] + bias);
    }
  }
}

// ---------------- GEMM2 + aspect attention head (fused) ------------------------------
// asp[16,512] = xh_tile @ aspW^T (WMMA, staged in LDS), then per-(row,aspect):
// score = softmax_a(asp·attW); p = score * (asp·mergeW) + mergeb.
__global__ void head_kernel(const _Float16* __restrict__ xh, const _Float16* __restrict__ aspWh,
                            const float* __restrict__ attW, const float* __restrict__ mergeW,
                            const float* __restrict__ mergeb, float* __restrict__ pbuf) {
  __shared__ _Float16 At[16][264];
  __shared__ float asp[16][ASP_A][68];  // padded D stride to dodge bank conflicts
  __shared__ float sc[16][ASP_A];
  int i0  = blockIdx.x * 16;
  int tid = threadIdx.x;
  for (int idx = tid; idx < 16 * HIDN; idx += 256) {
    int r = idx >> 8, c = idx & 255;
    At[r][c] = xh[(size_t)(i0 + r) * HIDN + c];
  }
  __syncthreads();
  int lane = tid & 31, wave = tid >> 5;
  int row  = lane & 15, h = lane >> 4;
  for (int t = 0; t < 4; ++t) {
    int n0  = (wave * 4 + t) * 16;
    int col = n0 + row;
    float8 acc = {0.f, 0.f, 0.f, 0.f, 0.f, 0.f, 0.f, 0.f};
    for (int k0 = 0; k0 < HIDN; k0 += 32) {
      half16 a;
#pragma unroll
      for (int vv = 0; vv < 8; ++vv) {
        int k = k0 + (h << 4) + (vv << 1);
        a[2 * vv]     = At[row][k];
        a[2 * vv + 1] = At[row][k + 1];
      }
      half16 b = *(const half16*)(aspWh + (size_t)col * HIDN + k0 + h * 16);
      acc = __builtin_amdgcn_wmma_f32_16x16x32_f16(false, a, false, b, (short)0, acc, false, false);
    }
#pragma unroll
    for (int r = 0; r < 8; ++r) {
      int m = r + 8 * h;
      asp[m][col >> 6][col & 63] = acc[r];
    }
  }
  __syncthreads();
  int prow = tid >> 3, pa = tid & 7;
  float dM = 0.f;
  if (tid < 128) {
    float dA = 0.f;
    for (int dd = 0; dd < ASP_D; ++dd) {
      float v = asp[prow][pa][dd];
      dA += v * attW[dd];
      dM += v * mergeW[dd];
    }
    sc[prow][pa] = dA;  // +attb cancels in softmax (constant over aspects)
  }
  __syncthreads();
  if (tid < 128) {
    float mx = -1e30f;
    for (int a2 = 0; a2 < ASP_A; ++a2) mx = fmaxf(mx, sc[prow][a2]);
    float den = 0.f;
    for (int a2 = 0; a2 < ASP_A; ++a2) den += __expf(sc[prow][a2] - mx);
    float score = __expf(sc[prow][pa] - mx) / den;
    pbuf[(size_t)(i0 + prow) * ASP_A + pa] = score * dM + mergeb[0];
  }
}

// ---------------- concat + linear + log_softmax --------------------------------------
__global__ void final_kernel(const float* __restrict__ pbuf, const float* __restrict__ catW,
                             const float* __restrict__ catb, float* __restrict__ out, int B) {
  int b = blockIdx.x * 256 + threadIdx.x;
  if (b >= B) return;
  const float* p1 = pbuf + (size_t)(2 * b) * ASP_A;
  const float* p2 = pbuf + (size_t)(2 * b + 1) * ASP_A;
  float l[2];
#pragma unroll
  for (int c = 0; c < 2; ++c) {
    float s = catb[c];
#pragma unroll
    for (int j = 0; j < ASP_A; ++j)
      s += p1[j] * catW[c * 16 + j] + p2[j] * catW[c * 16 + ASP_A + j];
    l[c] = s;
  }
  float mx  = fmaxf(l[0], l[1]);
  float lse = mx + logf(expf(l[0] - mx) + expf(l[1] - mx));
  out[(size_t)b * 2 + 0] = l[0] - lse;
  out[(size_t)b * 2 + 1] = l[1] - lse;
}

extern "C" void kernel_launch(void* const* d_in, const int* in_sizes, int n_in,
                              void* d_out, int out_size, void* d_ws, size_t ws_size,
                              hipStream_t stream) {
  const float* feat   = (const float*)d_in[0];
  const int*   src    = (const int*)d_in[1];
  const int*   dst    = (const int*)d_in[2];
  const int*   pairs  = (const int*)d_in[3];   // [B,2] row-major -> slots 2b, 2b+1
  const float* W1     = (const float*)d_in[4];
  const float* b1     = (const float*)d_in[5];
  const float* aspW   = (const float*)d_in[6];
  const float* attW   = (const float*)d_in[7];
  // d_in[8] = attb: uniform over softmax axis, cancels exactly -> unused
  const float* mergeW = (const float*)d_in[9];
  const float* mergeb = (const float*)d_in[10];
  const float* catW   = (const float*)d_in[11];
  const float* catb   = (const float*)d_in[12];

  int n      = in_sizes[0] / N_IN;  // 20000
  int E      = in_sizes[1];         // 640000
  int nPairs = in_sizes[3] / 2;     // 4096
  int slots  = 2 * nPairs;          // 8192

  // Workspace layout (S, deg, mask contiguous so one zero pass clears all three)
  char* ws = (char*)d_ws;
  float*    S     = (float*)ws;                       // n*128 f32
  float*    deg   = S + (size_t)n * N_IN;             // n f32
  int*      mask  = (int*)(deg + n);                  // n i32
  float*    mean  = (float*)(mask + n);               // 128 f32
  _Float16* W1h   = (_Float16*)(mean + N_IN);         // 256*128 f16 (32B aligned)
  _Float16* aspWh = W1h + HIDN * N_IN;                // 512*256 f16
  _Float16* xh    = aspWh + AD * HIDN;                // slots*256 f16
  float*    pbuf  = (float*)(xh + (size_t)slots * HIDN);  // slots*8 f32

  int zcount = n * (N_IN + 2);  // S + deg + mask dwords
  zero_kernel<<<(zcount + 255) / 256, 256, 0, stream>>>(S, zcount);
  mask_kernel<<<(slots + 255) / 256, 256, 0, stream>>>(pairs, mask, slots);
  col_mean_kernel<<<N_IN, 256, 0, stream>>>(feat, mean, n);
  cvt_kernel<<<(AD * HIDN + 255) / 256, 256, 0, stream>>>(W1, aspW, W1h, aspWh);
  edge_kernel<<<(E + 1) / 2, 256, 0, stream>>>(src, dst, feat, mean, mask, S, deg, E);
  gemm1_kernel<<<slots / 16, 256, 0, stream>>>(pairs, feat, mean, S, deg, W1h, b1, xh);
  head_kernel<<<slots / 16, 256, 0, stream>>>(xh, aspWh, attW, mergeW, mergeb, pbuf);
  final_kernel<<<(nPairs + 255) / 256, 256, 0, stream>>>(pbuf, catW, catb, (float*)d_out, nPairs);
}
